// DynamicFeedbackAttention_83391085019394
// MI455X (gfx1250) — matrix operand order repaired and checked by reading, not compiled, not run on hardware
//
#include <hip/hip_runtime.h>

// Problem constants (match reference)
#define Bc  4
#define Sc  1024
#define Dc  1024
#define Hc  16
#define DKc 64

typedef __attribute__((ext_vector_type(16))) __bf16 v16bf;
typedef __attribute__((ext_vector_type(8)))  __bf16 v8bf;
typedef __attribute__((ext_vector_type(8)))  float  v8f;

static __device__ __forceinline__ __bf16 f2bf(float f) {
  unsigned u = __builtin_bit_cast(unsigned, f);
  unsigned short s = (unsigned short)((u + 0x7FFFu + ((u >> 16) & 1u)) >> 16);  // RNE
  return __builtin_bit_cast(__bf16, s);
}

static __device__ __forceinline__ v16bf cat16(v8bf lo, v8bf hi) {
  v16bf a;
#pragma unroll
  for (int j = 0; j < 8; ++j) { a[j] = lo[j]; a[j + 8] = hi[j]; }
  return a;
}

// A fragment, 16x32 bf16, A row-major [M][K]:
// lane16 = M row; j0..7 -> K = kbase + hf*8 + j ; j8..15 -> K = kbase + 16 + hf*8 + (j-8)
static __device__ __forceinline__ v16bf load_a(const __bf16* A, int lda, size_t row0,
                                               int kbase, int lane16, int hf) {
  const __bf16* p = A + (row0 + (size_t)lane16) * (size_t)lda + kbase + hf * 8;
  v8bf lo = *(const v8bf*)p;
  v8bf hi = *(const v8bf*)(p + 16);
  return cat16(lo, hi);
}

// B fragment, 32x16 bf16, stored transposed as BT[N][K]:
// lane16 = N col; j0..7 -> K = kbase + hf*8 + j ; j8..15 -> K = kbase + 16 + hf*8 + (j-8)
static __device__ __forceinline__ v16bf load_b(const __bf16* BT, int ldb, size_t n0,
                                               int kbase, int lane16, int hf) {
  const __bf16* p = BT + (n0 + (size_t)lane16) * (size_t)ldb + kbase + hf * 8;
  v8bf lo = *(const v8bf*)p;
  v8bf hi = *(const v8bf*)(p + 16);
  return cat16(lo, hi);
}

static __device__ __forceinline__ v8f wmma_bf16(v16bf a, v16bf b, v8f c) {
  return __builtin_amdgcn_wmma_f32_16x16x32_bf16(false, a, false, b, (short)0, c, false, false);
}

// ---------------- prep kernels ----------------

__global__ void __launch_bounds__(256) cvt_kernel(const float* __restrict__ in,
                                                  __bf16* __restrict__ out, int n) {
  int i = blockIdx.x * 256 + threadIdx.x;
  if (i < n) out[i] = f2bf(in[i]);
}

// W [H, D, DK] f32 -> WT [H, DK, D] bf16   (N-major, K-contiguous for load_b)
__global__ void __launch_bounds__(256) twT_kernel(const float* __restrict__ W,
                                                  __bf16* __restrict__ WT) {
  int i = blockIdx.x * 256 + threadIdx.x;   // exact grid: H*D*DK / 256
  int h = i >> 16;                           // D*DK = 65536
  int rem = i & 65535;
  int d = rem >> 6;
  int k = rem & 63;
  WT[((size_t)h * DKc + k) * Dc + d] = f2bf(W[i]);
}

// ---------------- projection GEMM (64x16 tile per wave, 4x M-blocked) ----------------
// Explicit 2-deep ping-pong pipeline: while WMMAs consume fragment set A (k), fragment
// set B (k+32) is in flight; no register-rotation moves, steady s_wait_loadcnt ~10.

__global__ void __launch_bounds__(256) proj_kernel(const __bf16* __restrict__ A,
                                                   const __bf16* __restrict__ WT,
                                                   const float* __restrict__ bias,
                                                   __bf16* __restrict__ out,
                                                   int transposed) {
  int lane = threadIdx.x & 31, wid = threadIdx.x >> 5;
  int gw = blockIdx.x * 8 + wid;                    // 4096 waves total
  int tile = gw & 63;                               // 64 tiles per (b,h): 16 mb x 4 nt
  int bh = gw >> 6;
  int h = bh & (Hc - 1);
  int nt = tile & 3, mb = tile >> 2;
  int m0 = mb * 64, n0 = nt * 16;
  int lane16 = lane & 15, hf = lane >> 4;
  size_t arow = (size_t)(bh >> 4) * Sc + m0;        // b*S + m0
  const __bf16* Wh = WT + (size_t)h * DKc * Dc;

  v8f acc[4] = {};
  // prologue: fragment sets for k = 0 and k = 32
  v16bf wb0 = load_b(Wh, Dc, n0, 0, lane16, hf);
  v16bf aa0[4];
#pragma unroll
  for (int mi = 0; mi < 4; ++mi) aa0[mi] = load_a(A, Dc, arow + mi * 16, 0, lane16, hf);
  v16bf wb1 = load_b(Wh, Dc, n0, 32, lane16, hf);
  v16bf aa1[4];
#pragma unroll
  for (int mi = 0; mi < 4; ++mi) aa1[mi] = load_a(A, Dc, arow + mi * 16, 32, lane16, hf);

  for (int k = 64; k < Dc; k += 64) {
#pragma unroll
    for (int mi = 0; mi < 4; ++mi) acc[mi] = wmma_bf16(aa0[mi], wb0, acc[mi]);
    wb0 = load_b(Wh, Dc, n0, k, lane16, hf);
#pragma unroll
    for (int mi = 0; mi < 4; ++mi) aa0[mi] = load_a(A, Dc, arow + mi * 16, k, lane16, hf);
#pragma unroll
    for (int mi = 0; mi < 4; ++mi) acc[mi] = wmma_bf16(aa1[mi], wb1, acc[mi]);
    wb1 = load_b(Wh, Dc, n0, k + 32, lane16, hf);
#pragma unroll
    for (int mi = 0; mi < 4; ++mi) aa1[mi] = load_a(A, Dc, arow + mi * 16, k + 32, lane16, hf);
  }
#pragma unroll
  for (int mi = 0; mi < 4; ++mi) acc[mi] = wmma_bf16(aa0[mi], wb0, acc[mi]);
#pragma unroll
  for (int mi = 0; mi < 4; ++mi) acc[mi] = wmma_bf16(aa1[mi], wb1, acc[mi]);

  float bval = bias[h * DKc + n0 + lane16];
#pragma unroll
  for (int mi = 0; mi < 4; ++mi) {
#pragma unroll
    for (int r = 0; r < 8; ++r) {
      float vv = acc[mi][r] + bval;
      int srow = m0 + mi * 16 + r + hf * 8;
      if (transposed)   // V stored [B,H,DK,S]  (N-major for the context GEMM)
        out[((size_t)bh * DKc + n0 + lane16) * Sc + srow] = f2bf(vv);
      else              // Q/K stored [B,H,S,DK]
        out[((size_t)bh * Sc + srow) * DKc + n0 + lane16] = f2bf(vv);
    }
  }
}

// ---------------- fused flash attention (64-column key blocks) ----------------
// One wave handles a 16-row query block of one (b,h); 16 iterations over 64-key blocks.
// All K fragments are issued before the score WMMAs; all V fragments are issued before
// the softmax so their latency hides behind the VALU reduction work.

__global__ void __launch_bounds__(256) attn_kernel(const __bf16* __restrict__ Q,
                                                   const __bf16* __restrict__ Kb,
                                                   const __bf16* __restrict__ VT,
                                                   const int* __restrict__ msk,
                                                   const float* __restrict__ scaling,
                                                   __bf16* __restrict__ ctx) {
  __shared__ __bf16 Pl[8][16][64];                  // per-wave P staging, 16 KB
  int lane = threadIdx.x & 31, wid = threadIdx.x >> 5;
  int gw = blockIdx.x * 8 + wid;                    // 4096 waves total
  int mt = gw & 63;
  int bh = gw >> 6;
  int h = bh & (Hc - 1);
  int b = bh >> 4;
  int lane16 = lane & 15, hf = lane >> 4;
  int m0 = mt * 16;

  const __bf16* Qp = Q + (size_t)bh * Sc * DKc;
  const __bf16* Kp = Kb + (size_t)bh * Sc * DKc;
  const __bf16* Vp = VT + (size_t)bh * DKc * Sc;
  const int* mrow = msk + b * Sc;

  v16bf qa0 = load_a(Qp, DKc, m0, 0, lane16, hf);   // Q rows, K=0..31
  v16bf qa1 = load_a(Qp, DKc, m0, 32, lane16, hf);  // Q rows, K=32..63

  v8f o[4] = {};
  float mr[8], lr[8];
#pragma unroll
  for (int r = 0; r < 8; ++r) { mr[r] = -3.0e38f; lr[r] = 0.0f; }
  const float isq = 0.125f;                          // 1/sqrt(64)

  for (int t0 = 0; t0 < Sc; t0 += 64) {
    // ---- issue all 16 K-fragment loads up front ----
    v16bf kf[8];
    int mk[4];
#pragma unroll
    for (int st = 0; st < 4; ++st) {
      kf[st * 2]     = load_b(Kp, DKc, t0 + st * 16, 0,  lane16, hf);
      kf[st * 2 + 1] = load_b(Kp, DKc, t0 + st * 16, 32, lane16, hf);
      mk[st] = mrow[t0 + st * 16 + lane16];
    }
    __builtin_prefetch(Kp + (size_t)(t0 + 64 + lane16) * DKc, 0, 1);

    // ---- scores: 4 subtiles of 16 key positions, DK=64 in two k-chunks ----
    v8f s[4];
#pragma unroll
    for (int st = 0; st < 4; ++st) {
      v8f acc = {};
      acc = wmma_bf16(qa0, kf[st * 2],     acc);
      acc = wmma_bf16(qa1, kf[st * 2 + 1], acc);
      s[st] = acc;
    }

    // ---- issue all 16 V-fragment loads now; softmax below hides their latency ----
    v16bf vf[8];
#pragma unroll
    for (int nt = 0; nt < 4; ++nt) {
      vf[nt * 2]     = load_b(Vp, Sc, nt * 16, t0,      lane16, hf);
      vf[nt * 2 + 1] = load_b(Vp, Sc, nt * 16, t0 + 32, lane16, hf);
    }

    // ---- masked online softmax over 64 columns ----
#pragma unroll
    for (int r = 0; r < 8; ++r) {
      float f0 = mk[0] ? s[0][r] * isq : -9.0e15f;
      float f1 = mk[1] ? s[1][r] * isq : -9.0e15f;
      float f2 = mk[2] ? s[2][r] * isq : -9.0e15f;
      float f3 = mk[3] ? s[3][r] * isq : -9.0e15f;
      float mx = fmaxf(fmaxf(f0, f1), fmaxf(f2, f3));
      mx = fmaxf(mx, __shfl_xor(mx, 1, 16));
      mx = fmaxf(mx, __shfl_xor(mx, 2, 16));
      mx = fmaxf(mx, __shfl_xor(mx, 4, 16));
      mx = fmaxf(mx, __shfl_xor(mx, 8, 16));
      float mnew = fmaxf(mr[r], mx);
      float p0 = __expf(f0 - mnew);
      float p1 = __expf(f1 - mnew);
      float p2 = __expf(f2 - mnew);
      float p3 = __expf(f3 - mnew);
      float sm = (p0 + p1) + (p2 + p3);
      sm += __shfl_xor(sm, 1, 16);
      sm += __shfl_xor(sm, 2, 16);
      sm += __shfl_xor(sm, 4, 16);
      sm += __shfl_xor(sm, 8, 16);
      float alpha = __expf(mr[r] - mnew);
      lr[r] = lr[r] * alpha + sm;
      mr[r] = mnew;
#pragma unroll
      for (int nt = 0; nt < 4; ++nt) o[nt][r] *= alpha;
      // scatter P tile (C layout: row M = r + hf*8, cols st*16 + lane16)
      Pl[wid][r + hf * 8][lane16]      = f2bf(p0);
      Pl[wid][r + hf * 8][lane16 + 16] = f2bf(p1);
      Pl[wid][r + hf * 8][lane16 + 32] = f2bf(p2);
      Pl[wid][r + hf * 8][lane16 + 48] = f2bf(p3);
    }
    // cross-lane LDS RAW inside one wave: compiler can't see it, wait explicitly
    asm volatile("s_wait_dscnt 0" ::: "memory");

    // reload P as two 16x32 A fragments (k-chunks t0..+31 and t0+32..+63)
    const __bf16* prow = &Pl[wid][lane16][0];
    v16bf pa0 = cat16(*(const v8bf*)(prow + hf * 8),      *(const v8bf*)(prow + 16 + hf * 8));
    v16bf pa1 = cat16(*(const v8bf*)(prow + 32 + hf * 8), *(const v8bf*)(prow + 48 + hf * 8));

    // ---- context accumulation: O += P @ V (V fragments already resident) ----
#pragma unroll
    for (int nt = 0; nt < 4; ++nt) {
      o[nt] = wmma_bf16(pa0, vf[nt * 2],     o[nt]);
      o[nt] = wmma_bf16(pa1, vf[nt * 2 + 1], o[nt]);
    }
  }

  float sc = scaling[h];
#pragma unroll
  for (int r = 0; r < 8; ++r) {
    float f = sc / lr[r];
    size_t idx = ((size_t)b * Sc + m0 + r + hf * 8) * Dc + h * DKc + lane16;
#pragma unroll
    for (int nt = 0; nt < 4; ++nt)
      ctx[idx + nt * 16] = f2bf(o[nt][r] * f);
  }
}

// ---------------- output projection (64x16 tile per wave, ping-pong pipelined) ----------------

__global__ void __launch_bounds__(256) outproj_kernel(const __bf16* __restrict__ A,
                                                      const __bf16* __restrict__ WT, // Wout [N=dout][K=d]
                                                      const float* __restrict__ bias,
                                                      float* __restrict__ outf,
                                                      __bf16* __restrict__ outb) {
  int lane = threadIdx.x & 31, wid = threadIdx.x >> 5;
  int gw = blockIdx.x * 8 + wid;                    // 4096 waves
  int ntile = gw & 63;                              // 64 n-tiles
  int mb = (gw >> 6) & 15;                          // 16 m-blocks of 64 rows
  int b = gw >> 10;
  int m0 = mb * 64, n0 = ntile * 16;
  int lane16 = lane & 15, hf = lane >> 4;
  size_t arow = (size_t)b * Sc + m0;

  v8f acc[4] = {};
  v16bf wb0 = load_b(WT, Dc, n0, 0, lane16, hf);
  v16bf aa0[4];
#pragma unroll
  for (int mi = 0; mi < 4; ++mi) aa0[mi] = load_a(A, Dc, arow + mi * 16, 0, lane16, hf);
  v16bf wb1 = load_b(WT, Dc, n0, 32, lane16, hf);
  v16bf aa1[4];
#pragma unroll
  for (int mi = 0; mi < 4; ++mi) aa1[mi] = load_a(A, Dc, arow + mi * 16, 32, lane16, hf);

  for (int k = 64; k < Dc; k += 64) {
#pragma unroll
    for (int mi = 0; mi < 4; ++mi) acc[mi] = wmma_bf16(aa0[mi], wb0, acc[mi]);
    wb0 = load_b(WT, Dc, n0, k, lane16, hf);
#pragma unroll
    for (int mi = 0; mi < 4; ++mi) aa0[mi] = load_a(A, Dc, arow + mi * 16, k, lane16, hf);
#pragma unroll
    for (int mi = 0; mi < 4; ++mi) acc[mi] = wmma_bf16(aa1[mi], wb1, acc[mi]);
    wb1 = load_b(WT, Dc, n0, k + 32, lane16, hf);
#pragma unroll
    for (int mi = 0; mi < 4; ++mi) aa1[mi] = load_a(A, Dc, arow + mi * 16, k + 32, lane16, hf);
  }
#pragma unroll
  for (int mi = 0; mi < 4; ++mi) acc[mi] = wmma_bf16(aa0[mi], wb0, acc[mi]);
#pragma unroll
  for (int mi = 0; mi < 4; ++mi) acc[mi] = wmma_bf16(aa1[mi], wb1, acc[mi]);

  float bval = bias[n0 + lane16];
#pragma unroll
  for (int mi = 0; mi < 4; ++mi) {
#pragma unroll
    for (int r = 0; r < 8; ++r) {
      float vv = acc[mi][r] + bval;
      size_t idx = ((size_t)b * Sc + m0 + mi * 16 + r + hf * 8) * Dc + n0 + lane16;
      outf[idx] = vv;
      outb[idx] = f2bf(vv);
    }
  }
}

// ---------------- host orchestration ----------------

extern "C" void kernel_launch(void* const* d_in, const int* in_sizes, int n_in,
                              void* d_out, int out_size, void* d_ws, size_t ws_size,
                              hipStream_t stream) {
  (void)in_sizes; (void)n_in; (void)out_size; (void)ws_size;
  const float* x       = (const float*)d_in[0];
  const int*   mask    = (const int*)d_in[1];
  const float* Wq      = (const float*)d_in[2];
  const float* bq      = (const float*)d_in[3];
  const float* Wk      = (const float*)d_in[4];
  const float* bk      = (const float*)d_in[5];
  const float* Wv      = (const float*)d_in[6];
  const float* bv      = (const float*)d_in[7];
  const float* Wout    = (const float*)d_in[8];
  const float* bout    = (const float*)d_in[9];
  const float* scaling = (const float*)d_in[10];
  float* out = (float*)d_out;

  // workspace layout (~56 MB, 256B-aligned slices)
  char* ws = (char*)d_ws;
  size_t off = 0;
  auto take = [&](size_t bytes) -> char* {
    char* p = ws + off;
    off += (bytes + 255) & ~(size_t)255;
    return p;
  };
  const size_t NX = (size_t)Bc * Sc * Dc;        // 4M elems
  const size_t NP = (size_t)Bc * Hc * Sc * DKc;  // 4M elems
  const size_t NW = (size_t)Hc * Dc * DKc;       // 1M elems
  __bf16* xbf    = (__bf16*)take(NX * 2);
  __bf16* outbf  = (__bf16*)take(NX * 2);
  __bf16* ctxbf  = (__bf16*)take(NX * 2);
  __bf16* qbf    = (__bf16*)take(NP * 2);
  __bf16* kbf    = (__bf16*)take(NP * 2);
  __bf16* vT     = (__bf16*)take(NP * 2);
  __bf16* wqT    = (__bf16*)take(NW * 2);
  __bf16* wkT    = (__bf16*)take(NW * 2);
  __bf16* wvT    = (__bf16*)take(NW * 2);
  __bf16* woutbf = (__bf16*)take((size_t)Dc * Dc * 2);

  // prep
  cvt_kernel<<<(int)(NX / 256), 256, 0, stream>>>(x, xbf, (int)NX);
  cvt_kernel<<<(Dc * Dc) / 256, 256, 0, stream>>>(Wout, woutbf, Dc * Dc);
  twT_kernel<<<(int)(NW / 256), 256, 0, stream>>>(Wq, wqT);
  twT_kernel<<<(int)(NW / 256), 256, 0, stream>>>(Wk, wkT);
  twT_kernel<<<(int)(NW / 256), 256, 0, stream>>>(Wv, wvT);

  const int projBlocks = (Bc * Hc * (Sc / 64) * (DKc / 16)) / 8;  // 512
  const int attnBlocks = (Bc * Hc * (Sc / 16)) / 8;               // 512
  const int outBlocks  = (Bc * (Sc / 64) * (Dc / 16)) / 8;        // 512

  // K / V depend only on x: compute once
  proj_kernel<<<projBlocks, 256, 0, stream>>>(xbf, wkT, bk, kbf, 0);
  proj_kernel<<<projBlocks, 256, 0, stream>>>(xbf, wvT, bv, vT, 1);

  for (int it = 0; it < 3; ++it) {
    proj_kernel<<<projBlocks, 256, 0, stream>>>(it == 0 ? xbf : outbf, wqT, bq, qbf, 0);
    attn_kernel<<<attnBlocks, 256, 0, stream>>>(qbf, kbf, vT, mask, scaling, ctxbf);
    outproj_kernel<<<outBlocks, 256, 0, stream>>>(ctxbf, woutbf, bout, out, outbf);
  }
}
